// MRR_17420387353202
// MI455X (gfx1250) — matrix operand order repaired and checked by reading, not compiled, not run on hardware
//
#include <hip/hip_runtime.h>
#include <hip/hip_bf16.h>
#include <math.h>

typedef __attribute__((ext_vector_type(16))) __bf16 v16bf;
typedef __attribute__((ext_vector_type(8)))  float  v8f;
typedef __attribute__((ext_vector_type(4)))  int    v4i;

#define EPS_MRR 1e-8f
#define DM 128           // d_model
#define TPB 512          // 16 waves of 32

// ---- CDNA5 async global->LDS staging (hedged: fall back to sync copy) ------
#if defined(__HIP_DEVICE_COMPILE__) && \
    __has_builtin(__builtin_amdgcn_global_load_async_to_lds_b128)
#define MRR_ASYNC_LDS 1
typedef __attribute__((address_space(1))) v4i gv4i_t;   // global int4
typedef __attribute__((address_space(3))) v4i lv4i_t;   // LDS int4
#endif

__device__ __forceinline__ void mrr_async_wait_all()
{
#if defined(MRR_ASYNC_LDS)
#if __has_builtin(__builtin_amdgcn_s_wait_asynccnt)
    __builtin_amdgcn_s_wait_asynccnt(0);
#else
    asm volatile("s_wait_asynccnt 0x0" ::: "memory");
#endif
#endif
}

__device__ __forceinline__ float mrr_fast_sqrt(float x)
{
#if defined(__HIP_DEVICE_COMPILE__) && __has_builtin(__builtin_amdgcn_sqrtf)
    return __builtin_amdgcn_sqrtf(x);   // raw v_sqrt_f32, ~1 ulp
#else
    return sqrtf(x);
#endif
}

// ---------------------------------------------------------------------------
// Kernel 1: per-query norms + ground-truth similarity thresholds, zero counts
// ---------------------------------------------------------------------------
__global__ void mrr_prep_kernel(const float* __restrict__ yhat,
                                const float* __restrict__ pv,
                                const int*   __restrict__ gt,
                                float* __restrict__ t_thr,
                                float* __restrict__ ynorm,
                                int*   __restrict__ counts,
                                int batch)
{
    int b = blockIdx.x * blockDim.x + threadIdx.x;
    if (b >= batch) return;
    counts[b] = 0;

    const float4* yq = (const float4*)(yhat + (size_t)b * DM);
    const float4* pq = (const float4*)(pv + (size_t)gt[b] * DM);
    float syy = 0.f, spp = 0.f, dot = 0.f;
    #pragma unroll 4
    for (int i = 0; i < DM / 4; ++i) {
        float4 y = yq[i], p = pq[i];
        syy += y.x*y.x + y.y*y.y + y.z*y.z + y.w*y.w;
        spp += p.x*p.x + p.y*p.y + p.z*p.z + p.w*p.w;
        dot += y.x*p.x + y.y*p.y + y.z*p.z + y.w*p.w;
    }
    float yn = sqrtf(syy);
    ynorm[b] = yn;
    t_thr[b] = dot / fmaxf(yn * sqrtf(spp), EPS_MRR);  // gt cosine sim
}

// ---------------------------------------------------------------------------
// Kernel 2: streaming GEMM (bf16 hi/lo split WMMA) + count(sim > gt_sim)
// ---------------------------------------------------------------------------
__global__ __launch_bounds__(TPB, 1)
void mrr_main_kernel(const float* __restrict__ pv,
                     const float* __restrict__ yhat,
                     const float* __restrict__ t_thr,
                     const float* __restrict__ ynorm,
                     int* __restrict__ counts,
                     int ntiles)
{
    // double-buffered product tile (16 products x 128 dims)
    __shared__ float4 raw4[2][TPB];            // 2 x 8KB raw fp32
    __shared__ v16bf  bh[2][4][32];            // B operand, hi part, per K-chunk
    __shared__ v16bf  bl[2][4][32];            // B operand, lo part
    __shared__ float  pn2p[2][4][16];          // per-chunk |p|^2 partials

    const int tid  = threadIdx.x;
    const int wave = tid >> 5;                 // 0..15 : query tile id
    const int lane = tid & 31;
    const int m    = lane & 15;                // row/col index within 16
    const int g    = lane >> 4;                // half-wave group (K/M split)

    // ---- A operands: this wave's 16 queries, bf16 hi/lo, register resident.
    // A k order used here (contiguous, same as B): lane (m,g): elements[0..15]=K(16g..16g+15).
    v16bf a_hi[4], a_lo[4];
    {
        const float* yq = yhat + (size_t)(wave * 16 + m) * DM;
        #pragma unroll
        for (int kk = 0; kk < 4; ++kk) {
            v16bf h, l;
            #pragma unroll
            for (int part = 0; part < 2; ++part) {
                const int d0 = kk * 32 + g * 16 + part * 8;
                #pragma unroll
                for (int e = 0; e < 8; ++e) {
                    float  x  = yq[d0 + e];
                    __bf16 xh = (__bf16)x;
                    h[part * 8 + e] = xh;
                    l[part * 8 + e] = (__bf16)(x - (float)xh);
                }
            }
            a_hi[kk] = h; a_lo[kk] = l;
        }
    }

    // per-lane thresholds / query norms: C VGPR j holds row M = j + 8g
    float tj[8], yn[8];
    #pragma unroll
    for (int j = 0; j < 8; ++j) {
        int q = wave * 16 + 8 * g + j;
        tj[j] = t_thr[q];
        yn[j] = ynorm[q];
    }

    int acc[8];
    #pragma unroll
    for (int j = 0; j < 8; ++j) acc[j] = 0;

    const int stride = gridDim.x;
    const int tile0  = blockIdx.x;

    // stage one 8KB product tile into LDS (16B per thread, async when possible)
    auto stage = [&](int tile, int buf) {
        const float4* src = (const float4*)(pv + (size_t)tile * (16 * DM));
#if defined(MRR_ASYNC_LDS)
        __builtin_amdgcn_global_load_async_to_lds_b128(
            (gv4i_t*)(src + tid),
            (lv4i_t*)&raw4[buf][tid],
            0, 0);
#else
        raw4[buf][tid] = src[tid];
#endif
        int pf = tile + 2 * stride;
        if (pf < ntiles)
            __builtin_prefetch(pv + (size_t)pf * (16 * DM) + tid * 4, 0, 3);
    };

    if (tile0 < ntiles) stage(tile0, 0);

    int it = 0;
    for (int tile = tile0; tile < ntiles; tile += stride, ++it) {
        const int buf = it & 1;
        mrr_async_wait_all();                  // this wave's async stores landed
        __syncthreads();                       // raw[buf] visible to all waves

        // ---- convert fp32 -> bf16 hi/lo in WMMA B layout; waves 0..3, kk=wave
        // B k order used here (contiguous, same as A): lane (m,g): N=m, elements[0..15]=K(16g..16g+15)
        if (wave < 4) {
            const int    kk = wave;
            const float* rp = ((const float*)raw4[buf]) + m * DM + kk * 32 + g * 16;
            v16bf h, l;
            float s = 0.f;
            #pragma unroll
            for (int e = 0; e < 16; ++e) {
                float  x  = rp[e];
                s += x * x;
                __bf16 xh = (__bf16)x;
                h[e] = xh;
                l[e] = (__bf16)(x - (float)xh);
            }
            bh[buf][kk][lane] = h;
            bl[buf][kk][lane] = l;
            s += __shfl_xor(s, 16, 32);        // combine the two K-halves
            if (g == 0) pn2p[buf][kk][m] = s;
        }
        __syncthreads();                       // bh/bl/pn2p ready

        int next = tile + stride;              // overlap next tile's staging
        if (next < ntiles) stage(next, buf ^ 1);

        // ---- 16 queries x 16 products, K=128: 4 chunks x 3 split-WMMAs
        v8f c = {};
        #pragma unroll
        for (int kk = 0; kk < 4; ++kk) {
            v16bf bvh = bh[buf][kk][lane];
            v16bf bvl = bl[buf][kk][lane];
            c = __builtin_amdgcn_wmma_f32_16x16x32_bf16(false, a_hi[kk], false, bvh,
                                                        (short)0, c, false, false);
            c = __builtin_amdgcn_wmma_f32_16x16x32_bf16(false, a_hi[kk], false, bvl,
                                                        (short)0, c, false, false);
            c = __builtin_amdgcn_wmma_f32_16x16x32_bf16(false, a_lo[kk], false, bvh,
                                                        (short)0, c, false, false);
        }

        // product norm for this lane's column
        float pn2 = pn2p[buf][0][m] + pn2p[buf][1][m]
                  + pn2p[buf][2][m] + pn2p[buf][3][m];
        float pn = mrr_fast_sqrt(pn2);

        // sim > gt_sim  <=>  dot > t * max(|y||p|, eps)
        #pragma unroll
        for (int j = 0; j < 8; ++j) {
            float denom = fmaxf(yn[j] * pn, EPS_MRR);
            acc[j] += (c[j] > tj[j] * denom) ? 1 : 0;
        }
    }

    // reduce per-query hit counts over the 16 product lanes, flush once
    #pragma unroll
    for (int j = 0; j < 8; ++j) {
        int v = acc[j];
        v += __shfl_xor(v, 8, 32);
        v += __shfl_xor(v, 4, 32);
        v += __shfl_xor(v, 2, 32);
        v += __shfl_xor(v, 1, 32);
        if (m == 0) atomicAdd(&counts[wave * 16 + 8 * g + j], v);
    }
}

// ---------------------------------------------------------------------------
// Kernel 3: MRR = mean(1 / (1 + count_b))
// ---------------------------------------------------------------------------
__global__ void mrr_final_kernel(const int* __restrict__ counts,
                                 float* __restrict__ out, int batch)
{
    __shared__ float red[256];
    int tid = threadIdx.x;
    red[tid] = (tid < batch) ? 1.0f / (float)(1 + counts[tid]) : 0.f;
    __syncthreads();
    for (int s = 128; s > 0; s >>= 1) {
        if (tid < s) red[tid] += red[tid + s];
        __syncthreads();
    }
    if (tid == 0) out[0] = red[0] / (float)batch;
}

// ---------------------------------------------------------------------------
extern "C" void kernel_launch(void* const* d_in, const int* in_sizes, int n_in,
                              void* d_out, int out_size, void* d_ws, size_t ws_size,
                              hipStream_t stream)
{
    const float* yhat = (const float*)d_in[0];
    const float* pv   = (const float*)d_in[1];
    const int*   gt   = (const int*)d_in[2];
    float*       out  = (float*)d_out;

    const int       batch = in_sizes[0] / DM;                 // 256
    const long long np    = (long long)in_sizes[1] / DM;      // 1,000,000
    const int       ntiles = (int)(np / 16);                  // 62,500

    float* t_thr  = (float*)d_ws;
    float* ynorm  = t_thr + 256;
    int*   counts = (int*)(ynorm + 256);

    mrr_prep_kernel<<<1, 256, 0, stream>>>(yhat, pv, gt, t_thr, ynorm, counts, batch);

    int grid = ntiles < 1024 ? ntiles : 1024;
    mrr_main_kernel<<<grid, TPB, 0, stream>>>(pv, yhat, t_thr, ynorm, counts, ntiles);

    mrr_final_kernel<<<1, 256, 0, stream>>>(counts, out, batch);
}
